// GemLiteMatmul_4870492913665
// MI455X (gfx1250) — compile-verified, matrix operand order checked by reading, not run on hardware
//
#include <hip/hip_runtime.h>

typedef _Float16 half_t;
typedef __attribute__((ext_vector_type(16))) _Float16 v16h;
typedef __attribute__((ext_vector_type(2)))  _Float16 h2;
typedef __attribute__((ext_vector_type(8)))  float    v8f;
typedef __attribute__((ext_vector_type(4)))  unsigned int u32x4;

#define KTOT   8192
#define NTOT   8192
#define MOUT   8
#define SUPERK 1024          // K elements staged in LDS per round
#define LDSPAD 8             // halves of pad per row -> 2-way-max ds_load_b128 banks

union Frag {
    v16h         h;
    u32x4        q[2];
    unsigned int u[8];
    h2           p[8];
};

struct WW { unsigned int w[16]; };   // one 256-K group of packed words for this lane

// Packed layout: word = row*1024 + 32*(K>>8) + (K&31); nibble level i=(K>>5)&7
// at bits [28-4i..]. Lane (h=lane>>4) needs words {g*32+h*8+0..7} and
// {g*32+16+h*8+0..7}: 4 x non-temporal B128 loads per 256-K group, every nibble
// consumed (the wave covers 16 rows x 128B = whole cachelines).
__global__ __launch_bounds__(256) void gemlite_w4a16_wmma(
    const half_t* __restrict__ x,
    const unsigned int* __restrict__ Wq,
    const half_t* __restrict__ scales,
    const half_t* __restrict__ zeros,
    float* __restrict__ part,        // [S][8][N] fp32 partials (S>1)
    half_t* __restrict__ out,        // direct fp16 out (S==1)
    int S)
{
    __shared__ half_t xs[16][SUPERK + LDSPAD];

    const int Kc   = KTOT / S;
    const int sIdx = blockIdx.y;
    const int tid  = threadIdx.x;
    const int wave = tid >> 5;
    const int lane = tid & 31;
    const int h    = lane >> 4;       // K-half of the WMMA fragment
    const int nr   = lane & 15;       // N column (B/D) == M row (A)
    const int n    = blockIdx.x * 128 + wave * 16 + nr;

    // per-column bias (1024 + zero_point): 0x6400|c == f16(1024+c)
    const half_t bh = (half_t)(1024.0f + (float)zeros[n]);
    h2 bias2 = { bh, bh };
    const unsigned int K64 = 0x64646464u;

    v8f acc = {0.f, 0.f, 0.f, 0.f, 0.f, 0.f, 0.f, 0.f};

    const unsigned int* wrow = Wq + (size_t)n * (KTOT / 8) + h * 8;

    auto load_w = [&](int g) {       // g = global 256-K group index
        const unsigned int* wr = wrow + g * 32;
        u32x4 v0 = __builtin_nontemporal_load((const u32x4*)(wr));
        u32x4 v1 = __builtin_nontemporal_load((const u32x4*)(wr + 4));
        u32x4 v2 = __builtin_nontemporal_load((const u32x4*)(wr + 16));
        u32x4 v3 = __builtin_nontemporal_load((const u32x4*)(wr + 20));
        WW r;
#pragma unroll
        for (int q = 0; q < 4; ++q) {
            r.w[q] = v0[q]; r.w[4 + q] = v1[q];
            r.w[8 + q] = v2[q]; r.w[12 + q] = v3[q];
        }
        return r;
    };

    const int kBeg = sIdx * Kc;
    for (int kk = kBeg; kk < kBeg + Kc; kk += SUPERK) {
        // ---- stage x[0..7][kk..kk+SUPERK) into LDS; zero padded rows 8..15 ----
        __syncthreads();
        for (int it = tid; it < 2048; it += 256) {     // 16 rows * 128 uint4
            const int r = it >> 7;
            const int c = it & 127;
            u32x4 v = {0u, 0u, 0u, 0u};
            if (r < MOUT)
                v = *(const u32x4*)(x + (size_t)r * KTOT + kk + c * 8);
            *(u32x4*)(&xs[r][c * 8]) = v;
        }
        __syncthreads();

        const int g0 = kk >> 8;
        WW wcur = load_w(g0);                           // W pipeline prologue

        for (int gl = 0; gl < SUPERK / 256; ++gl) {
            WW wnxt;
            const bool more = (gl + 1 < SUPERK / 256);
            if (more) wnxt = load_w(g0 + gl + 1);       // overlap with compute

            // --- byte-lane nibble split: e = levels {7,5,3,1}, o = {6,4,2,0} ---
            unsigned int me1[8], me2[8], mo1[8], mo2[8];
#pragma unroll
            for (int p = 0; p < 8; ++p) {
                unsigned int w0 = wcur.w[2 * p], w1 = wcur.w[2 * p + 1];
                unsigned int e0 = w0 & 0x0F0F0F0Fu;
                unsigned int e1 = w1 & 0x0F0F0F0Fu;
                unsigned int o0 = (w0 >> 4) & 0x0F0F0F0Fu;
                unsigned int o1 = (w1 >> 4) & 0x0F0F0F0Fu;
                // {even.b0, odd.b0, even.b1, odd.b1} / {even.b2, odd.b2, ...}
                me1[p] = __builtin_amdgcn_perm(e0, e1, 0x01050004u); // lvl 7 | 5
                me2[p] = __builtin_amdgcn_perm(e0, e1, 0x03070206u); // lvl 3 | 1
                mo1[p] = __builtin_amdgcn_perm(o0, o1, 0x01050004u); // lvl 6 | 4
                mo2[p] = __builtin_amdgcn_perm(o0, o1, 0x03070206u); // lvl 2 | 0
            }

            const int klocal = gl * 256;
            auto load_a = [&](int i) {                 // A fragment for level i
                Frag a;
                const int kl = klocal + i * 32 + h * 8;
                a.q[0] = *(const u32x4*)(&xs[nr][kl]);
                a.q[1] = *(const u32x4*)(&xs[nr][kl + 16]);
                return a;
            };
            auto build_b = [&](const unsigned int* m, unsigned int sel) {
                Frag b;
#pragma unroll
                for (int p = 0; p < 8; ++p) {
                    // -> 0x64oo64ee = f16x2(1024+c_even, 1024+c_odd)
                    unsigned int raw = __builtin_amdgcn_perm(m[p], K64, sel);
                    b.p[p] = __builtin_bit_cast(h2, raw) - bias2;  // exact c - z
                }
                return b;
            };
#define SEL_LO 0x00050004u   // bytes {m.b0, 0x64, m.b1, 0x64}
#define SEL_HI 0x00070006u   // bytes {m.b2, 0x64, m.b3, 0x64}

            // levels ordered so each m-array dies in two steps; A ping-pong so
            // the ds_load for the next level is in flight across each WMMA
            Frag a_cur = load_a(7), a_nxt, b;
            b = build_b(me1, SEL_LO); a_nxt = load_a(5);
            acc = __builtin_amdgcn_wmma_f32_16x16x32_f16(false, a_cur.h, false, b.h, (short)0, acc, false, false);
            b = build_b(me1, SEL_HI); a_cur = load_a(3);
            acc = __builtin_amdgcn_wmma_f32_16x16x32_f16(false, a_nxt.h, false, b.h, (short)0, acc, false, false);
            b = build_b(me2, SEL_LO); a_nxt = load_a(1);
            acc = __builtin_amdgcn_wmma_f32_16x16x32_f16(false, a_cur.h, false, b.h, (short)0, acc, false, false);
            b = build_b(me2, SEL_HI); a_cur = load_a(6);
            acc = __builtin_amdgcn_wmma_f32_16x16x32_f16(false, a_nxt.h, false, b.h, (short)0, acc, false, false);
            b = build_b(mo1, SEL_LO); a_nxt = load_a(4);
            acc = __builtin_amdgcn_wmma_f32_16x16x32_f16(false, a_cur.h, false, b.h, (short)0, acc, false, false);
            b = build_b(mo1, SEL_HI); a_cur = load_a(2);
            acc = __builtin_amdgcn_wmma_f32_16x16x32_f16(false, a_nxt.h, false, b.h, (short)0, acc, false, false);
            b = build_b(mo2, SEL_LO); a_nxt = load_a(0);
            acc = __builtin_amdgcn_wmma_f32_16x16x32_f16(false, a_cur.h, false, b.h, (short)0, acc, false, false);
            b = build_b(mo2, SEL_HI);
            acc = __builtin_amdgcn_wmma_f32_16x16x32_f16(false, a_nxt.h, false, b.h, (short)0, acc, false, false);

            if (more) wcur = wnxt;
        }
    }

    // D layout: VGPR r, lanes 0-15 -> (M=r, N=lane). Only M<8 is real data.
    if (h == 0) {
        if (S == 1) {
            const float scl = (float)scales[n];
#pragma unroll
            for (int r = 0; r < 8; ++r)
                out[(size_t)r * NTOT + n] = (half_t)(scl * acc[r]);
        } else {
#pragma unroll
            for (int r = 0; r < 8; ++r)
                part[((size_t)(sIdx * 8 + r)) * NTOT + n] = acc[r];
        }
    }
}

__global__ __launch_bounds__(256) void gemlite_reduce(
    const float* __restrict__ part,
    const half_t* __restrict__ scales,
    half_t* __restrict__ out, int S)
{
    const int t = blockIdx.x * 256 + threadIdx.x;   // [0, 8*8192)
    const int n = t & (NTOT - 1);
    const int m = t >> 13;
    float a = 0.f;
    for (int s2 = 0; s2 < S; ++s2)
        a += part[((size_t)(s2 * 8 + m)) * NTOT + n];
    out[t] = (half_t)((float)scales[n] * a);
}

extern "C" void kernel_launch(void* const* d_in, const int* in_sizes, int n_in,
                              void* d_out, int out_size, void* d_ws, size_t ws_size,
                              hipStream_t stream)
{
    (void)in_sizes; (void)n_in; (void)out_size;
    const half_t*       x      = (const half_t*)d_in[0];
    const half_t*       scales = (const half_t*)d_in[1];
    const half_t*       zeros  = (const half_t*)d_in[2];
    const unsigned int* Wq     = (const unsigned int*)d_in[3];
    // d_in[4] = W_nbits (==4), baked into the kernel
    half_t* out = (half_t*)d_out;
    float*  ws  = (float*)d_ws;

    // split-K for occupancy (4096 waves at S=8); deterministic two-pass reduce
    int S = 1;
    if      (ws_size >= (size_t)8 * MOUT * NTOT * sizeof(float)) S = 8;
    else if (ws_size >= (size_t)4 * MOUT * NTOT * sizeof(float)) S = 4;
    else if (ws_size >= (size_t)2 * MOUT * NTOT * sizeof(float)) S = 2;

    dim3 grid(NTOT / 128, S);
    gemlite_w4a16_wmma<<<grid, dim3(256), 0, stream>>>(
        x, Wq, scales, zeros, (S > 1) ? ws : nullptr, out, S);
    if (S > 1)
        gemlite_reduce<<<(MOUT * NTOT) / 256, dim3(256), 0, stream>>>(ws, scales, out, S);
}
